// MambaEncoder_2705829396807
// MI455X (gfx1250) — matrix operand order repaired and checked by reading, not compiled, not run on hardware
//
#include <hip/hip_runtime.h>
#include <hip/hip_bf16.h>

// ---------------------------------------------------------------------------
// CDNA5 (gfx1250) Mamba-encoder kernels.
// GEMM work -> V_WMMA_F32_16X16X32_F16 with fragment-major pre-swizzled
// operands (zero guards / zero shuffles in the inner loop), A staged through
// LDS with async global->LDS copies (double buffered), B read from L2.
// Scan / norm / conv -> VALU streaming (bandwidth-bound on 23.3 TB/s HBM).
// ---------------------------------------------------------------------------

typedef __attribute__((ext_vector_type(16))) _Float16 v16h;
typedef __attribute__((ext_vector_type(8)))  float    v8f;
typedef int v4i_vs __attribute__((vector_size(16)));   // matches builtin param

#define GLOBAL_AS __attribute__((address_space(1)))
#define LDS_AS    __attribute__((address_space(3)))

#define D_STATE 16
#define D_CONV 4

#if defined(__has_builtin)
#  if __has_builtin(__builtin_amdgcn_global_load_async_to_lds_b128)
#    define HAVE_ASYNC_LDS 1
#  endif
#  if __has_builtin(__builtin_amdgcn_s_wait_asynccnt)
#    define HAVE_WAIT_ASYNC 1
#  endif
#endif
#ifndef HAVE_ASYNC_LDS
#  define HAVE_ASYNC_LDS 0
#endif
#ifndef HAVE_WAIT_ASYNC
#  define HAVE_WAIT_ASYNC 0
#endif

__device__ __forceinline__ void wait_asynccnt0() {
#if HAVE_WAIT_ASYNC
  __builtin_amdgcn_s_wait_asynccnt(0);
#else
  asm volatile("s_wait_asynccnt 0x0" ::: "memory");
#endif
}

// ---------------------------------------------------------------------------
// Fragment-major packing (f32 -> f16), padded with zeros to tile multiples.
//
// A fragment block (mb = 16-row tile, kt = 32-K tile): 32 lanes x 16 halves,
// per-lane contiguous, in the ISA 7.12.2 16-bit A 16x32 order:
//   lanes 0-15 : K 0..7 then 16..23 ; lanes 16-31 : K 8..15 then 24..31
// ---------------------------------------------------------------------------
__global__ void pack_a_frag(const float* __restrict__ A, _Float16* __restrict__ Ap,
                            int M, int K, int MT, int KT)
{
  size_t i = (size_t)blockIdx.x * blockDim.x + threadIdx.x;
  size_t tot = (size_t)MT * KT * 512;
  if (i >= tot) return;
  int e    = (int)(i & 15);
  int lane = (int)((i >> 4) & 31);
  size_t blk = i >> 9;              // mb*KT + kt
  int kt = (int)(blk % KT);
  int mb = (int)(blk / KT);
  int m  = mb * 16 + (lane & 15);
  int kh = (lane >> 4) * 8;
  int k  = kt * 32 + ((e < 8) ? (kh + e) : (16 + kh + (e - 8)));
  _Float16 v = (_Float16)0.0f;
  if (m < M && k < K) v = (_Float16)A[(size_t)m * K + k];
  Ap[i] = v;
}

// B fragment block (nb = 16-col tile, kt): lane l -> n = nb*16 + (l&15),
// elements e=0..15 -> same K order as A: lanes 0-15 K 0..7,16..23; lanes 16-31 K 8..15,24..31.
__global__ void pack_b_frag(const float* __restrict__ B, _Float16* __restrict__ Bp,
                            int K, int N, int NT, int KT)
{
  size_t i = (size_t)blockIdx.x * blockDim.x + threadIdx.x;
  size_t tot = (size_t)NT * KT * 512;
  if (i >= tot) return;
  int e    = (int)(i & 15);
  int lane = (int)((i >> 4) & 31);
  size_t blk = i >> 9;              // nb*KT + kt
  int kt = (int)(blk % KT);
  int nb = (int)(blk / KT);
  int n  = nb * 16 + (lane & 15);
  int k  = kt * 32 + ((e < 8) ? ((lane >> 4) * 8 + e) : (16 + (lane >> 4) * 8 + (e - 8)));
  _Float16 v = (_Float16)0.0f;
  if (k < K && n < N) v = (_Float16)B[(size_t)k * N + n];
  Bp[i] = v;
}

// ---------------------------------------------------------------------------
// WMMA GEMM on fragment-major operands.
// Block: 256 threads = 8 waves -> block tile 64(M) x 64(N).
// Wave (mWi = w&3, nWi = w>>2): one 16-row tile, two 16-col tiles
// (2 accumulators reuse one A fragment). K stepped by 32.
// A fragments: async global->LDS double buffer (or direct global fallback).
// B fragments: direct global (weights are L2-resident; 192MB L2).
// act: 0=none 1=relu 2=gelu(erf) 3=silu
// ---------------------------------------------------------------------------
__global__ __launch_bounds__(256) void wmma_gemm_f16(
    const _Float16* __restrict__ Ap, const _Float16* __restrict__ Bp,
    const float* __restrict__ bias, float* __restrict__ C,
    int M, int N, int K, int KT, int act)
{
  const int tid  = threadIdx.x;
  const int wave = tid >> 5;
  const int lane = tid & 31;
  const int mWi  = wave & 3;
  const int nWi  = wave >> 2;

  const size_t mb  = (size_t)blockIdx.x * 4 + mWi;        // 16-row tile
  const size_t nb0 = (size_t)blockIdx.y * 4 + nWi * 2;    // 16-col tiles
  const size_t nb1 = nb0 + 1;

  v8f acc0 = {};
  v8f acc1 = {};

#if HAVE_ASYNC_LDS
  // 2 stages x 4 m-tiles x (32 lanes x 16 halves) = 2 x 4KB
  __shared__ __attribute__((aligned(32))) _Float16 As[2][4 * 512];
  const size_t mb0 = (size_t)blockIdx.x * 4;
  // thread t copies one 16B chunk: m-tile = t>>6, chunk = t&63
  const int cpTile  = tid >> 6;
  const int cpChunk = tid & 63;
  const _Float16* cpSrcBase =
      Ap + ((mb0 + cpTile) * KT) * 512 + (size_t)cpChunk * 8;

  __builtin_amdgcn_global_load_async_to_lds_b128(
      (GLOBAL_AS v4i_vs*)cpSrcBase,                              // kt = 0
      (LDS_AS v4i_vs*)&As[0][cpTile * 512 + cpChunk * 8], 0, 0);

  int stage = 0;
  for (int kt = 0; kt < KT; ++kt) {
    wait_asynccnt0();
    __syncthreads();
    if (kt + 1 < KT) {
      __builtin_amdgcn_global_load_async_to_lds_b128(
          (GLOBAL_AS v4i_vs*)(cpSrcBase + (size_t)(kt + 1) * 512),
          (LDS_AS v4i_vs*)&As[stage ^ 1][cpTile * 512 + cpChunk * 8], 0, 0);
      __builtin_prefetch(Bp + ((nb0 * KT) + kt + 1) * 512 + lane * 16, 0, 1);
    }
    v16h a  = *(const v16h*)(&As[stage][mWi * 512 + lane * 16]);  // ds_load_b128 x2
    v16h b0 = *(const v16h*)(Bp + (nb0 * KT + kt) * 512 + lane * 16);
    v16h b1 = *(const v16h*)(Bp + (nb1 * KT + kt) * 512 + lane * 16);
    acc0 = __builtin_amdgcn_wmma_f32_16x16x32_f16(
        false, a, false, b0, (short)0, acc0, false, false);
    acc1 = __builtin_amdgcn_wmma_f32_16x16x32_f16(
        false, a, false, b1, (short)0, acc1, false, false);
    __syncthreads();
    stage ^= 1;
  }
#else
  for (int kt = 0; kt < KT; ++kt) {
    if (kt + 1 < KT)
      __builtin_prefetch(Ap + (mb * KT + kt + 1) * 512 + lane * 16, 0, 1);
    v16h a  = *(const v16h*)(Ap + (mb * KT + kt) * 512 + lane * 16);
    v16h b0 = *(const v16h*)(Bp + (nb0 * KT + kt) * 512 + lane * 16);
    v16h b1 = *(const v16h*)(Bp + (nb1 * KT + kt) * 512 + lane * 16);
    acc0 = __builtin_amdgcn_wmma_f32_16x16x32_f16(
        false, a, false, b0, (short)0, acc0, false, false);
    acc1 = __builtin_amdgcn_wmma_f32_16x16x32_f16(
        false, a, false, b1, (short)0, acc1, false, false);
  }
#endif

  // epilogue: C/D layout -> VGPR i : M = mBase + (lane>=16 ? 8:0) + i, N = lane&15
  const int mBase = (int)mb * 16 + (lane >> 4) * 8;
#pragma unroll
  for (int j = 0; j < 2; ++j) {
    const v8f acc = j ? acc1 : acc0;
    const int n = (int)(j ? nb1 : nb0) * 16 + (lane & 15);
    const float bv = (bias && n < N) ? bias[n] : 0.0f;
#pragma unroll
    for (int i = 0; i < 8; ++i) {
      int m = mBase + i;
      if (m < M && n < N) {
        float v = acc[i] + bv;
        if      (act == 1) v = fmaxf(v, 0.0f);
        else if (act == 2) v = 0.5f * v * (1.0f + erff(v * 0.70710678f));
        else if (act == 3) v = v / (1.0f + __expf(-v));
        C[(size_t)m * N + n] = v;
      }
    }
  }
  __builtin_amdgcn_s_wait_tensorcnt(0);
}

// ---------------------------------------------------------------------------
// Direct 3-D convolution (stem / down / GSC 3x3x3). Thread per output element.
// ---------------------------------------------------------------------------
__global__ void conv3d_kernel(const float* __restrict__ x,
                              const float* __restrict__ w,
                              const float* __restrict__ b,
                              float* __restrict__ y,
                              int Ci, int Co, int Di, int Hi, int Wi,
                              int Do, int Ho, int Wo,
                              int ks, int stride, int pad, int relu)
{
  long idx = (long)blockIdx.x * blockDim.x + threadIdx.x;
  long tot = (long)Co * Do * Ho * Wo;
  if (idx >= tot) return;
  int wo = idx % Wo;  long t = idx / Wo;
  int ho = t % Ho;    t /= Ho;
  int dd = t % Do;    int co = t / Do;

  float acc = b ? b[co] : 0.0f;
  for (int ci = 0; ci < Ci; ++ci) {
    const float* xc = x + (size_t)ci * Di * Hi * Wi;
    const float* wc = w + ((size_t)co * Ci + ci) * ks * ks * ks;
    for (int kz = 0; kz < ks; ++kz) {
      int iz = dd * stride - pad + kz;
      if (iz < 0 || iz >= Di) continue;
      for (int ky = 0; ky < ks; ++ky) {
        int iy = ho * stride - pad + ky;
        if (iy < 0 || iy >= Hi) continue;
        for (int kx = 0; kx < ks; ++kx) {
          int ix = wo * stride - pad + kx;
          if (ix < 0 || ix >= Wi) continue;
          acc += wc[(kz * ks + ky) * ks + kx] *
                 xc[((size_t)iz * Hi + iy) * Wi + ix];
        }
      }
    }
  }
  if (relu) acc = fmaxf(acc, 0.0f);
  y[idx] = acc;
}

// ---------------------------------------------------------------------------
// Instance/feature norm (+ optional ReLU). Block per channel, LDS reduction.
// ---------------------------------------------------------------------------
__global__ __launch_bounds__(256) void inorm_act(const float* __restrict__ x,
                                                 float* __restrict__ y,
                                                 int S, int relu)
{
  int c = blockIdx.x;
  const float* xc = x + (size_t)c * S;
  __shared__ float rs[256], rq[256];
  float s = 0.f, q = 0.f;
  for (int i = threadIdx.x; i < S; i += 256) {
    float v = xc[i]; s += v; q += v * v;
  }
  rs[threadIdx.x] = s; rq[threadIdx.x] = q;
  __syncthreads();
  for (int off = 128; off > 0; off >>= 1) {
    if ((int)threadIdx.x < off) {
      rs[threadIdx.x] += rs[threadIdx.x + off];
      rq[threadIdx.x] += rq[threadIdx.x + off];
    }
    __syncthreads();
  }
  float mean = rs[0] / S;
  float var  = rq[0] / S - mean * mean;
  float inv  = rsqrtf(var + 1e-5f);
  float* yc = y + (size_t)c * S;
  for (int i = threadIdx.x; i < S; i += 256) {
    float v = (xc[i] - mean) * inv;
    yc[i] = relu ? fmaxf(v, 0.0f) : v;
  }
}

// ---------------------------------------------------------------------------
// Depthwise causal conv1d + SiLU
// ---------------------------------------------------------------------------
__global__ void causal_conv_silu(const float* __restrict__ x,
                                 const float* __restrict__ w,
                                 const float* __restrict__ b,
                                 float* __restrict__ y, int D, int L)
{
  long idx = (long)blockIdx.x * blockDim.x + threadIdx.x;
  if (idx >= (long)D * L) return;
  int d = idx / L, l = idx % L;
  float acc = b[d];
#pragma unroll
  for (int k = 0; k < D_CONV; ++k) {
    int t = l - (D_CONV - 1) + k;
    if (t >= 0) acc += w[d * D_CONV + k] * x[(size_t)d * L + t];
  }
  y[idx] = acc / (1.0f + __expf(-acc));
}

// ---------------------------------------------------------------------------
// Selective scan: thread per inner channel d, sequential over L, 16 states.
// Streams u/dt/B/C at ~4N flop/elt -> HBM-bandwidth bound; h stays in VGPRs.
// ---------------------------------------------------------------------------
__global__ void selective_scan(const float* __restrict__ u,
                               const float* __restrict__ dtraw,
                               const float* __restrict__ dtb,
                               const float* __restrict__ Alog,
                               const float* __restrict__ Bm,   // [N][L]
                               const float* __restrict__ Cm,   // [N][L]
                               const float* __restrict__ Dv,
                               const float* __restrict__ z,
                               float* __restrict__ y,
                               int Dd, int L)
{
  int d = blockIdx.x * blockDim.x + threadIdx.x;
  if (d >= Dd) return;
  float h[D_STATE], A[D_STATE];
#pragma unroll
  for (int s = 0; s < D_STATE; ++s) {
    h[s] = 0.0f;
    A[s] = -__expf(Alog[(size_t)d * D_STATE + s]);
  }
  float Dc = Dv[d], bias = dtb[d];
  for (int t = 0; t < L; ++t) {
    float dr = dtraw[(size_t)d * L + t] + bias;
    float dt = (dr > 20.0f) ? dr : __logf(1.0f + __expf(dr));  // softplus
    float ut = u[(size_t)d * L + t];
    float du = dt * ut;
    float acc = 0.0f;
#pragma unroll
    for (int s = 0; s < D_STATE; ++s) {
      h[s] = __expf(dt * A[s]) * h[s] + du * Bm[(size_t)s * L + t];
      acc += h[s] * Cm[(size_t)s * L + t];
    }
    float yt = acc + Dc * ut;
    float zt = z[(size_t)d * L + t];
    y[(size_t)d * L + t] = yt * (zt / (1.0f + __expf(-zt)));
  }
}

// ---------------------------------------------------------------------------
// c = a + b (residual)
// ---------------------------------------------------------------------------
__global__ void add_kernel(const float* __restrict__ a,
                           const float* __restrict__ b,
                           float* __restrict__ c, size_t n) {
  size_t i = (size_t)blockIdx.x * blockDim.x + threadIdx.x;
  if (i < n) c[i] = a[i] + b[i];
}

// ---------------------------------------------------------------------------
// Orchestration
// ---------------------------------------------------------------------------
static inline int ceil_div(long a, long b) { return (int)((a + b - 1) / b); }
static inline int round_up(int a, int b) { return ((a + b - 1) / b) * b; }

extern "C" void kernel_launch(void* const* d_in, const int* in_sizes, int n_in,
                              void* d_out, int out_size, void* d_ws, size_t ws_size,
                              hipStream_t stream) {
  static const int dims[4] = {48, 96, 192, 384};

  const float* x = (const float*)d_in[0];
  float* out = (float*)d_out;

  // deterministic parameter cursor over the flattened param list
  int pc = 1;
  auto nextp = [&]() -> const float* {
    const float* p = (const float*)d_in[pc % (n_in > 0 ? n_in : 1)];
    pc++;
    return p;
  };

  // bump allocator over workspace
  size_t cur = 0;
  auto alloc = [&](size_t bytes) -> void* {
    void* p = (char*)d_ws + cur;
    cur += (bytes + 255) & ~(size_t)255;
    return p;
  };

  const size_t MAXL  = 27648;     // stage-0 sequence length (mult of 64)
  const size_t MAXDI = 2 * 384;
  float*    X   = (float*)   alloc(sizeof(float)    * 384 * MAXL);
  float*    T1  = (float*)   alloc(sizeof(float)    * MAXDI * MAXL);
  float*    T2  = (float*)   alloc(sizeof(float)    * MAXDI * MAXL);
  float*    XZ  = (float*)   alloc(sizeof(float)    * 2 * MAXDI * MAXL);
  float*    UB  = (float*)   alloc(sizeof(float)    * MAXDI * MAXL);
  float*    DBL = (float*)   alloc(sizeof(float)    * 64 * MAXL);
  float*    DT  = (float*)   alloc(sizeof(float)    * MAXDI * MAXL);
  float*    Y   = (float*)   alloc(sizeof(float)    * MAXDI * MAXL);
  _Float16* HA  = (_Float16*)alloc(sizeof(_Float16) * MAXDI * MAXL);  // A frags
  _Float16* HB  = (_Float16*)alloc(sizeof(_Float16) * 2048 * 1024);   // B frags
  (void)ws_size; (void)in_sizes; (void)out_size;

  // C[M,N] = act(A[M,K] x B[K,N] + bias); operands pre-swizzled + padded
  auto gemm = [&](const float* a32, const float* b32, const float* bias,
                  float* c, int M, int N, int K, int act) {
    const int Mp = round_up(M, 64), Np = round_up(N, 64), Kp = round_up(K, 32);
    const int MT = Mp / 16, NT = Np / 16, KT = Kp / 32;
    pack_a_frag<<<ceil_div((long)MT * KT * 512, 256), 256, 0, stream>>>(
        a32, HA, M, K, MT, KT);
    pack_b_frag<<<ceil_div((long)NT * KT * 512, 256), 256, 0, stream>>>(
        b32, HB, K, N, NT, KT);
    dim3 grid(Mp / 64, Np / 64);
    wmma_gemm_f16<<<grid, 256, 0, stream>>>(HA, HB, bias, c, M, N, K, KT, act);
  };

  int D = 48, H = 24, W = 24;   // spatial after stem
  size_t out_off = 0;

  for (int i = 0; i < 4; ++i) {
    const int C  = dims[i];
    const int di = 2 * C;
    const int dtr = (C + 15) / 16;

    if (i == 0) {
      // stem: 1 -> 48, 7^3, stride 2, pad 3 on the 96x48x48 input
      long tot = (long)C * D * H * W;
      conv3d_kernel<<<ceil_div(tot, 256), 256, 0, stream>>>(
          x, nextp(), nextp(), X, 1, C, 96, 48, 48, D, H, W, 7, 2, 3, 0);
    } else {
      // inorm + 2^3 stride-2 down conv
      int Ci = dims[i - 1];
      inorm_act<<<Ci, 256, 0, stream>>>(X, T1, D * H * W, 0);
      int Do = D / 2, Ho = H / 2, Wo = W / 2;
      long tot = (long)C * Do * Ho * Wo;
      conv3d_kernel<<<ceil_div(tot, 256), 256, 0, stream>>>(
          T1, nextp(), nextp(), X, Ci, C, D, H, W, Do, Ho, Wo, 2, 2, 0, 0);
      D = Do; H = Ho; W = Wo;
    }
    const int L = D * H * W;

    // ---- GSC (3x3x3 conv + inorm/relu + residual) ----
    {
      long tot = (long)C * L;
      conv3d_kernel<<<ceil_div(tot, 256), 256, 0, stream>>>(
          X, nextp(), nextp(), T1, C, C, D, H, W, D, H, W, 3, 1, 1, 0);
      inorm_act<<<C, 256, 0, stream>>>(T1, T2, L, 1);
      add_kernel<<<ceil_div(tot, 256), 256, 0, stream>>>(T2, X, X, (size_t)C * L);
    }

    // ---- 2 Mamba layers ----
    for (int layer = 0; layer < 2; ++layer) {
      inorm_act<<<C, 256, 0, stream>>>(X, T1, L, 0);                 // norm
      gemm(T1, nextp(), nullptr, XZ, L, 2 * di, C, 0);               // in_proj
      const float* xc = XZ;
      const float* z  = XZ + (size_t)di * L;
      causal_conv_silu<<<ceil_div((long)di * L, 256), 256, 0, stream>>>(
          xc, nextp(), nextp(), UB, di, L);                          // dw conv
      gemm(UB, nextp(), nullptr, DBL, L, dtr + 2 * D_STATE, di, 0);  // x_proj
      gemm(DBL, nextp(), nullptr, DT, L, di, dtr, 0);                // dt_proj
      const float* dtb  = nextp();
      const float* Alog = nextp();
      const float* Bm   = DBL + (size_t)dtr * L;
      const float* Cm   = DBL + (size_t)(dtr + D_STATE) * L;
      const float* Dv   = nextp();
      selective_scan<<<ceil_div(di, 128), 128, 0, stream>>>(
          UB, DT, dtb, Alog, Bm, Cm, Dv, z, Y, di, L);               // SSM scan
      gemm(Y, nextp(), nullptr, T1, L, C, di, 0);                    // out_proj
      add_kernel<<<ceil_div((long)C * L, 256), 256, 0, stream>>>(
          T1, X, X, (size_t)C * L);                                  // residual
    }

    // ---- stage-output MLP: inorm -> fc1(gelu) -> fc2 into d_out ----
    inorm_act<<<C, 256, 0, stream>>>(X, T1, L, 0);
    gemm(T1, nextp(), nextp(), T2, L, 2 * C, C, 2);                  // fc1+gelu
    gemm(T2, nextp(), nextp(), out + out_off, L, C, 2 * C, 0);       // fc2
    out_off += (size_t)C * L;
  }
}
